// RGCN_26877905338706
// MI455X (gfx1250) — matrix operand order repaired and checked by reading, not compiled, not run on hardware
//
#include <hip/hip_runtime.h>

// ---------------------------------------------------------------------------
// RGCN (3 layers, basis decomposition) for MI455X / gfx1250, wave32 + WMMA.
//
//  - bf16 activations & weights, f32 accumulate via v_wmma_f32_16x16x32_bf16
//  - A tile staged to LDS once per block via global_load_async_to_lds_b128
//    (ASYNCcnt path), fragments read via ds_load_b128, bank-conflict-free pad
//  - Wt stored as [17][dout_pad][din] so B fragments are contiguous 32B loads
//  - relation 16 == root weight; its GEMM output seeds agg = x@root + bias
//  - edge phase: vectorized bf16 gather, scale by 1/max(deg,1), f32 atomics
// ---------------------------------------------------------------------------

typedef __attribute__((ext_vector_type(16))) __bf16 v16bf;
typedef __attribute__((ext_vector_type(8)))  __bf16 v8bf;
typedef __attribute__((ext_vector_type(4)))  __bf16 v4bf;
typedef __attribute__((ext_vector_type(8)))  float  v8f;

#define N_NODES   100000
#define N_EDGES   1600000
#define NUM_RELS  16
#define NUM_BASES 8
#define DIN       128          // input dim of every layer
#define LDS_STRIDE 136         // 128 + 8 pad: 272B rows -> conflict-free ds_load_b128

// ---------------------------------------------------------------- utilities
__global__ void k_f32_to_bf16(const float* __restrict__ in,
                              __bf16* __restrict__ out, size_t n) {
    size_t i = (size_t)blockIdx.x * blockDim.x + threadIdx.x;
    if (i < n) out[i] = (__bf16)in[i];
}

__global__ void k_relu_to_bf16(const float* __restrict__ in,
                               __bf16* __restrict__ out, size_t n) {
    size_t i = (size_t)blockIdx.x * blockDim.x + threadIdx.x;
    if (i < n) {
        float v = in[i];
        out[i] = (__bf16)(v > 0.0f ? v : 0.0f);
    }
}

__global__ void k_zero_f32(float* __restrict__ p, size_t n) {
    size_t i = (size_t)blockIdx.x * blockDim.x + threadIdx.x;
    if (i < n) p[i] = 0.0f;
}

// degree: deg[dst*16 + rel] += 1   (identical for all three layers)
__global__ void k_deg_count(const int* __restrict__ edst,
                            const int* __restrict__ etype,
                            float* __restrict__ deg, int E) {
    int e = blockIdx.x * blockDim.x + threadIdx.x;
    if (e < E) atomicAdd(&deg[edst[e] * NUM_RELS + etype[e]], 1.0f);
}

// ------------------------------------------------------------- weight build
// Wt[r][o][i] = sum_b comp[r][b]*bases[b][i][o]   (r<16)
// Wt[16][o][i] = root[i][o]; zero-pad rows o in [dout, dop)
__global__ void k_build_wt(const float* __restrict__ bases,  // [8][din][dout]
                           const float* __restrict__ comp,   // [16][8]
                           const float* __restrict__ root,   // [din][dout]
                           __bf16* __restrict__ Wt,          // [17][dop][din]
                           int dout, int dop) {
    size_t idx = (size_t)blockIdx.x * blockDim.x + threadIdx.x;
    size_t tot = (size_t)17 * dop * DIN;
    if (idx >= tot) return;
    int i = (int)(idx % DIN);
    int o = (int)((idx / DIN) % dop);
    int r = (int)(idx / ((size_t)DIN * dop));
    float v = 0.0f;
    if (o < dout) {
        if (r < NUM_RELS) {
            #pragma unroll
            for (int b = 0; b < NUM_BASES; ++b)
                v += comp[r * NUM_BASES + b] *
                     bases[((size_t)b * DIN + i) * dout + o];
        } else {
            v = root[(size_t)i * dout + o];
        }
    }
    Wt[idx] = (__bf16)v;
}

// ---------------------------------------------------------------- WMMA GEMM
// grid: (N/16, 17), block: 32 * (dop/16) threads (one wave per 16-col tile).
// A tile (16 x 128 bf16) staged to LDS via async-to-LDS, then ds_load frags.
// r < 16 : xw[r][row][col] (bf16)
// r == 16: agg[row][col] = c + bias[col] (f32, only col < dout)
__global__ __launch_bounds__(256)
void k_rgcn_gemm_wmma(const __bf16* __restrict__ act,   // [N][DIN]
                      const __bf16* __restrict__ Wt,    // [17][dop][DIN]
                      __bf16* __restrict__ xw,          // [16][N][dop]
                      float*  __restrict__ agg,         // [N][dout]
                      const float* __restrict__ bias,   // [dout]
                      int N, int dout, int dop) {
    __shared__ __bf16 As[16 * LDS_STRIDE];

    const int lane    = threadIdx.x & 31;
    const int wave    = threadIdx.x >> 5;
    const int rowBase = blockIdx.x * 16;
    const int r       = blockIdx.y;            // 0..16
    const int colBase = wave * 16;
    const int half    = lane >> 4;             // 0 or 1
    const int m       = lane & 15;

    // ---- stage A tile: 16 rows x 128 cols bf16 = 256 chunks of 16B ----
    // async DMA into LDS, tracked with ASYNCcnt (gfx1250 path)
    const unsigned ldsBase = (unsigned)(uintptr_t)As;  // low 32b of flat = LDS offset
    for (int c = threadIdx.x; c < 256; c += blockDim.x) {
        int row = c >> 4;                 // 16 chunks (of 8 bf16) per row
        int ce  = (c & 15) * 8;           // element offset within row
        const __bf16* g = act + (size_t)(rowBase + row) * DIN + ce;
        unsigned l = ldsBase + (unsigned)(row * LDS_STRIDE + ce) * 2u;
        asm volatile("global_load_async_to_lds_b128 %0, %1, off"
                     :: "v"(l), "v"(g) : "memory");
    }
    asm volatile("s_wait_asynccnt 0x0" ::: "memory");
    __syncthreads();

    const __bf16* arow = As + m * LDS_STRIDE;                       // LDS
    const __bf16* brow = Wt + ((size_t)r * dop + (colBase + m)) * DIN;

    v8f c = {0.f, 0.f, 0.f, 0.f, 0.f, 0.f, 0.f, 0.f};

    #pragma unroll
    for (int k0 = 0; k0 < DIN; k0 += 32) {
        // A fragment (16x32 bf16): lanes 0-15 hold K {0..15} of row m,
        // lanes 16-31 hold K {16..31}  -> two 16B ds_load_b128.
        const int ka = k0 + half * 16;
        union { v16bf v; v8bf h[2]; } A;
        A.h[0] = *(const v8bf*)(arow + ka);
        A.h[1] = *(const v8bf*)(arow + ka + 8);
        // B fragment (32x16 bf16): lane holds column colBase+m, K-half
        // {0..15} or {16..31}; Wt is [col][k] so it's one 32B global load.
        const int kb = k0 + half * 16;
        v16bf B = *(const v16bf*)(brow + kb);

        c = __builtin_amdgcn_wmma_f32_16x16x32_bf16(
                false, A.v, false, B, (short)0, c, false, false);
    }

    // C/D layout: VGPR i holds row (i + 8*half), column = lane&15.
    if (r < NUM_RELS) {
        __bf16* out = xw + ((size_t)r * N + rowBase) * dop + colBase + m;
        #pragma unroll
        for (int i = 0; i < 8; ++i) {
            int row = i + 8 * half;
            out[(size_t)row * dop] = (__bf16)c[i];
        }
    } else {
        int col = colBase + m;
        if (col < dout) {
            float b = bias[col];
            #pragma unroll
            for (int i = 0; i < 8; ++i) {
                int row = rowBase + i + 8 * half;
                agg[(size_t)row * dout + col] = c[i] + b;
            }
        }
    }
}

// -------------------------------------------------------------- edge scatter
// one wave per edge: agg[dst] += xw[rel][src] * (1/max(deg,1))
// lane loads 4 contiguous bf16 (8B) -> 4 f32 atomics
__global__ __launch_bounds__(256)
void k_rgcn_scatter(const int* __restrict__ esrc,
                    const int* __restrict__ edst,
                    const int* __restrict__ etype,
                    const __bf16* __restrict__ xw,   // [16][N][dop]
                    const float* __restrict__ deg,   // [N][16]
                    float* __restrict__ agg,         // [N][dout]
                    int E, int N, int dout, int dop) {
    int e = blockIdx.x * (blockDim.x >> 5) + (threadIdx.x >> 5);
    if (e >= E) return;
    int lane = threadIdx.x & 31;
    int src = esrc[e], dst = edst[e], rel = etype[e];
    float norm = 1.0f / fmaxf(deg[dst * NUM_RELS + rel], 1.0f);
    const __bf16* row = xw + ((size_t)rel * N + src) * dop;
    float* orow = agg + (size_t)dst * dout;
    for (int f0 = lane * 4; f0 < dop; f0 += 128) {
        v4bf v = *(const v4bf*)(row + f0);
        #pragma unroll
        for (int j = 0; j < 4; ++j) {
            int f = f0 + j;
            if (f < dout) atomicAdd(orow + f, (float)v[j] * norm);
        }
    }
}

// ---------------------------------------------------------------------------
extern "C" void kernel_launch(void* const* d_in, const int* in_sizes, int n_in,
                              void* d_out, int out_size, void* d_ws, size_t ws_size,
                              hipStream_t stream) {
    const float* x     = (const float*)d_in[0];
    const int*   esrc  = (const int*)d_in[1];
    const int*   edst  = (const int*)d_in[2];
    const int*   etype = (const int*)d_in[3];
    const float* bases[3] = {(const float*)d_in[4], (const float*)d_in[8],  (const float*)d_in[12]};
    const float* comp [3] = {(const float*)d_in[5], (const float*)d_in[9],  (const float*)d_in[13]};
    const float* root [3] = {(const float*)d_in[6], (const float*)d_in[10], (const float*)d_in[14]};
    const float* bias [3] = {(const float*)d_in[7], (const float*)d_in[11], (const float*)d_in[15]};

    const int N = N_NODES, E = N_EDGES;
    const int D = 128, C = 40, Cp = 48;   // hidden, classes, padded classes

    // ---- workspace carve-up (bytes, 256-aligned) ----
    char* ws = (char*)d_ws;
    size_t off = 0;
    auto carve = [&](size_t bytes) {
        char* p = ws + off;
        off += (bytes + 255) & ~(size_t)255;
        return p;
    };
    __bf16* act0 = (__bf16*)carve((size_t)N * D * 2);            // 25.6 MB
    __bf16* act1 = (__bf16*)carve((size_t)N * D * 2);            // 25.6 MB
    __bf16* Wt   = (__bf16*)carve((size_t)17 * D * D * 2);       // 0.56 MB
    float*  deg  = (float*) carve((size_t)N * NUM_RELS * 4);     // 6.4 MB
    float*  agg  = (float*) carve((size_t)N * D * 4);            // 51.2 MB
    __bf16* xw   = (__bf16*)carve((size_t)NUM_RELS * N * D * 2); // 409.6 MB
    (void)ws_size;

    const size_t nxd = (size_t)N * D;

    // x -> bf16
    k_f32_to_bf16<<<(unsigned)((nxd + 255) / 256), 256, 0, stream>>>(x, act0, nxd);

    // per-(dst,rel) degree (shared by all layers)
    k_zero_f32<<<(unsigned)(((size_t)N * NUM_RELS + 255) / 256), 256, 0, stream>>>(
        deg, (size_t)N * NUM_RELS);
    k_deg_count<<<(E + 255) / 256, 256, 0, stream>>>(edst, etype, deg, E);

    dim3 gemmGrid(N / 16, 17);

    // ---------------- layer 0: 128 -> 128, relu ----------------
    k_build_wt<<<(unsigned)(((size_t)17 * D * D + 255) / 256), 256, 0, stream>>>(
        bases[0], comp[0], root[0], Wt, D, D);
    k_rgcn_gemm_wmma<<<gemmGrid, 32 * (D / 16), 0, stream>>>(
        act0, Wt, xw, agg, bias[0], N, D, D);
    k_rgcn_scatter<<<E / 8, 256, 0, stream>>>(
        esrc, edst, etype, xw, deg, agg, E, N, D, D);
    k_relu_to_bf16<<<(unsigned)((nxd + 255) / 256), 256, 0, stream>>>(agg, act1, nxd);

    // ---------------- layer 1: 128 -> 128, relu ----------------
    k_build_wt<<<(unsigned)(((size_t)17 * D * D + 255) / 256), 256, 0, stream>>>(
        bases[1], comp[1], root[1], Wt, D, D);
    k_rgcn_gemm_wmma<<<gemmGrid, 32 * (D / 16), 0, stream>>>(
        act1, Wt, xw, agg, bias[1], N, D, D);
    k_rgcn_scatter<<<E / 8, 256, 0, stream>>>(
        esrc, edst, etype, xw, deg, agg, E, N, D, D);
    k_relu_to_bf16<<<(unsigned)((nxd + 255) / 256), 256, 0, stream>>>(agg, act0, nxd);

    // ---------------- layer 2: 128 -> 40, no activation --------
    float* out = (float*)d_out;   // [N][40], seeded by root pass then scattered into
    k_build_wt<<<(unsigned)(((size_t)17 * Cp * D + 255) / 256), 256, 0, stream>>>(
        bases[2], comp[2], root[2], Wt, C, Cp);
    k_rgcn_gemm_wmma<<<gemmGrid, 32 * (Cp / 16), 0, stream>>>(
        act0, Wt, xw, out, bias[2], N, C, Cp);
    k_rgcn_scatter<<<E / 8, 256, 0, stream>>>(
        esrc, edst, etype, xw, deg, out, E, N, C, Cp);
}
